// FABlock2D_88493506166999
// MI455X (gfx1250) — compile-verified, matrix operand order checked by reading, not run on hardware
//
#include <hip/hip_runtime.h>
#include <hip/hip_bf16.h>

// ---------------------------------------------------------------------------
// FABlock2D for MI455X (gfx1250): all GEMMs via v_wmma_f32_16x16x32_bf16.
// B=4, NY=NX=128, DIM=256, HEADS=8, DH=64, HD=512, HID=1024, NTOK=65536.
// Double-buffered LDS staging; B tile stored transposed so both A and B
// fragments are two contiguous ds_load_b128 per lane (no scalar packing).
// ---------------------------------------------------------------------------

typedef __bf16 bfx8  __attribute__((ext_vector_type(8)));
typedef __bf16 bfx16 __attribute__((ext_vector_type(16)));
typedef float  fx8   __attribute__((ext_vector_type(8)));

__device__ __forceinline__ float gelu_f(float x) {
    float x3 = x * x * x;
    return 0.5f * x * (1.0f + tanhf(0.7978845608028654f * (x + 0.044715f * x3)));
}

// ------------------------------ generic WMMA GEMM ---------------------------
// C[M,N] = act(alpha * A[M,K] @ B[K,N] + bias) + residual, batched over grid.z
// with per-operand batch divisors (offset = (z / div) * stride).
// outmode: 0 = f32, 1 = bf16, 2 = bf16 with v-tensor scatter [b,h,y,x*64+c].
struct GemmP {
    const __bf16* A; long long strideA; int a_div; int lda;
    const __bf16* B; long long strideB; int b_div; int ldb;
    void* C;         long long strideC; int c_div; int ldc;
    const float* bias;
    const float* residual;   // f32, same (row,col) indexing as C with ldc
    int M, N, K;
    float alpha;
    int act;                 // 0 none, 1 tanh-gelu
    int outmode;
};

__global__ __launch_bounds__(256) void gemm_bf16(GemmP p) {
    // [buf][row][k] ; row stride 40 elems (80 B) keeps 16-byte alignment.
    __shared__ __bf16 As [2][128][40];   // A tile, row-major (m, k)
    __shared__ __bf16 Bst[2][128][40];   // B tile, transposed (n, k)

    const int tid  = threadIdx.x;
    const int wave = tid >> 5;
    const int lane = tid & 31;
    const int half = lane >> 4;      // 16-lane group
    const int r    = lane & 15;      // row (A/C) / col (B/C) within tile

    const int z = blockIdx.z;
    const __bf16* A = p.A + (long long)(z / p.a_div) * p.strideA;
    const __bf16* B = p.B + (long long)(z / p.b_div) * p.strideB;
    const int row0 = blockIdx.y * 128;
    const int col0 = blockIdx.x * 128;

    // A staging: 128 rows x 2 chunks of 16
    const int ar = tid >> 1, ac = (tid & 1) * 16;
    // B staging: 16 k-pairs x 16 col-groups of 8
    const int bk = (tid >> 4) * 2;
    const int bn = (tid & 15) * 8;
    const bool arow_ok = (row0 + ar < p.M);
    const bool bcol_ok = (col0 + bn < p.N);   // N is a multiple of 16

    bfx8 a0, a1, b0, b1;
    auto gload = [&](int k0) {
        bfx8 zz = {};
        a0 = zz; a1 = zz; b0 = zz; b1 = zz;
        if (arow_ok) {
            const bfx8* s = (const bfx8*)(A + (long long)(row0 + ar) * p.lda + (k0 + ac));
            a0 = s[0]; a1 = s[1];
        }
        if (bcol_ok) {
            const __bf16* s0 = B + (long long)(k0 + bk) * p.ldb + (col0 + bn);
            b0 = *(const bfx8*)s0;
            b1 = *(const bfx8*)(s0 + p.ldb);
        }
    };
    auto stage = [&](int buf) {
        *(bfx8*)(&As[buf][ar][ac])     = a0;
        *(bfx8*)(&As[buf][ar][ac + 8]) = a1;
        #pragma unroll
        for (int i = 0; i < 8; ++i) {
            __bf16* d = &Bst[buf][bn + i][bk];   // bk even -> dword aligned pair
            d[0] = b0[i];
            d[1] = b1[i];
        }
    };

    fx8 acc[8] = {};

    auto compute = [&](int buf) {
        // 16-bit A 16x32 fragment: lane r (group `half`) holds two contiguous
        // 8-element runs: K = half*8+0..7 and K = 16+half*8+0..7.
        const __bf16* arow = &As[buf][wave * 16 + r][0];
        bfx8 alo = *(const bfx8*)(arow + half * 8);
        bfx8 ahi = *(const bfx8*)(arow + 16 + half * 8);
        bfx16 afrag = __builtin_shufflevector(alo, ahi,
            0, 1, 2, 3, 4, 5, 6, 7, 8, 9, 10, 11, 12, 13, 14, 15);
        #pragma unroll
        for (int nt = 0; nt < 8; ++nt) {
            const __bf16* brow = &Bst[buf][nt * 16 + r][0];
            bfx8 blo = *(const bfx8*)(brow + half * 8);
            bfx8 bhi = *(const bfx8*)(brow + 16 + half * 8);
            bfx16 bfrag = __builtin_shufflevector(blo, bhi,
                0, 1, 2, 3, 4, 5, 6, 7, 8, 9, 10, 11, 12, 13, 14, 15);
            acc[nt] = __builtin_amdgcn_wmma_f32_16x16x32_bf16(
                false, afrag, false, bfrag, (short)0, acc[nt], false, false);
        }
    };

    const int nsteps = p.K >> 5;
    gload(0);
    stage(0);
    __syncthreads();
    for (int s = 0; s < nsteps; ++s) {
        const int cur = s & 1;
        const bool has_next = (s + 1 < nsteps);
        if (s + 2 < nsteps) {   // gfx1250 global_prefetch_b8 for the K+2 tile
            __builtin_prefetch(A + (long long)(row0 + ar) * p.lda + ((s + 2) * 32 + ac), 0, 0);
            __builtin_prefetch(B + (long long)((s + 2) * 32 + bk) * p.ldb + (col0 + bn), 0, 0);
        }
        if (has_next) gload((s + 1) * 32);
        compute(cur);
        if (has_next) stage(cur ^ 1);
        __syncthreads();
    }

    // Epilogue. C/D 16x16 f32 layout: VGPR j -> M = j + half*8, N = lane&15.
    long long cbase = (long long)(z / p.c_div) * p.strideC;
    const bool full = (row0 + 128 <= p.M) && (col0 + 128 <= p.N);
    #pragma unroll
    for (int nt = 0; nt < 8; ++nt) {
        #pragma unroll
        for (int j = 0; j < 8; ++j) {
            int gr = row0 + wave * 16 + half * 8 + j;
            int gc = col0 + nt * 16 + r;
            if (full || (gr < p.M && gc < p.N)) {
                float v = acc[nt][j] * p.alpha;
                if (p.bias)     v += p.bias[gc];
                if (p.act == 1) v = gelu_f(v);
                if (p.residual) v += p.residual[(long long)gr * p.ldc + gc];
                if (p.outmode == 0) {
                    ((float*)p.C)[cbase + (long long)gr * p.ldc + gc] = v;
                } else if (p.outmode == 1) {
                    ((__bf16*)p.C)[cbase + (long long)gr * p.ldc + gc] = (__bf16)v;
                } else {
                    // scatter v into [b,h,y, x*64+c] for einsum1
                    int b = gr >> 14, y = (gr >> 7) & 127, x = gr & 127;
                    int h = gc >> 6,  c = gc & 63;
                    long long idx = ((long long)((b * 8 + h) * 128 + y) * 8192) + x * 64 + c;
                    ((__bf16*)p.C)[idx] = (__bf16)v;
                }
            }
        }
    }
}

// ------------------------------ elementwise kernels -------------------------

__global__ void cvt_kernel(const float* __restrict__ in, __bf16* __restrict__ out, int n) {
    int i = blockIdx.x * blockDim.x + threadIdx.x;
    if (i < n) out[i] = (__bf16)in[i];
}

// LayerNorm over 256 features, one row per block.
__global__ __launch_bounds__(256) void ln_kernel(const float* __restrict__ x,
                                                 const float* __restrict__ g,
                                                 const float* __restrict__ b,
                                                 __bf16* __restrict__ out) {
    __shared__ float red[256];
    int row = blockIdx.x, t = threadIdx.x;
    float v = x[(long long)row * 256 + t];
    red[t] = v; __syncthreads();
    for (int o = 128; o > 0; o >>= 1) { if (t < o) red[t] += red[t + o]; __syncthreads(); }
    float m = red[0] * (1.0f / 256.0f); __syncthreads();
    float d = v - m;
    red[t] = d * d; __syncthreads();
    for (int o = 128; o > 0; o >>= 1) { if (t < o) red[t] += red[t + o]; __syncthreads(); }
    float var = red[0] * (1.0f / 256.0f);
    float inv = rsqrtf(var + 1e-5f);
    out[(long long)row * 256 + t] = (__bf16)(d * inv * g[t] + b[t]);
}

// Mean-pool un over x (axis==0 -> pool_y[b,y,d]) or over y (axis==1 -> pool_x[b,x,d]).
__global__ __launch_bounds__(256) void pool_kernel(const __bf16* __restrict__ un,
                                                   __bf16* __restrict__ out, int axis) {
    int bq = blockIdx.x;             // b*128 + q
    int b = bq >> 7, q = bq & 127;
    int d = threadIdx.x;
    float s = 0.0f;
    if (axis == 0) {
        const __bf16* base = un + ((long long)(b * 128 + q) * 128) * 256 + d;
        for (int x = 0; x < 128; ++x) s += (float)base[(long long)x * 256];
    } else {
        const __bf16* base = un + ((long long)(b * 128) * 128 + q) * 256 + d;
        for (int y = 0; y < 128; ++y) s += (float)base[(long long)y * 128 * 256];
    }
    out[(long long)bq * 256 + d] = (__bf16)(s * (1.0f / 128.0f));
}

// RoPE on qk[(b*128+n), 1024] -> q [bh][n][64] and k^T [bh][64][n] (bf16).
__global__ __launch_bounds__(512) void rope_kernel(const float* __restrict__ qk,
                                                   const float* __restrict__ cosb,
                                                   const float* __restrict__ sinb,
                                                   __bf16* __restrict__ qout,
                                                   __bf16* __restrict__ ktout) {
    int bn = blockIdx.x;             // b*128 + n
    int n  = bn & 127;
    int t  = threadIdx.x;            // h*64 + c
    int c  = t & 63;
    float cs = cosb[n * 64 + c], sn = sinb[n * 64 + c];
    long long base = (long long)bn * 1024;
    float q  = qk[base + t];
    float k  = qk[base + 512 + t];
    int   t2 = (c < 32) ? (t + 32) : (t - 32);
    float q2 = qk[base + t2];
    float k2 = qk[base + 512 + t2];
    float rq = (c < 32) ? -q2 : q2;      // rotate_half
    float rk = (c < 32) ? -k2 : k2;
    float qo = q * cs + rq * sn;
    float ko = k * cs + rk * sn;
    int h  = t >> 6;
    int bh = (bn >> 7) * 8 + h;
    qout [((long long)bh * 128 + n) * 64 + c]  = (__bf16)qo;
    ktout[((long long)bh * 64 + c) * 128 + n]  = (__bf16)ko;
}

// Row softmax over 128 columns.
__global__ __launch_bounds__(128) void softmax_kernel(const float* __restrict__ s,
                                                      __bf16* __restrict__ out) {
    __shared__ float red[128];
    int row = blockIdx.x, t = threadIdx.x;
    float v = s[(long long)row * 128 + t];
    red[t] = v; __syncthreads();
    for (int o = 64; o > 0; o >>= 1) { if (t < o) red[t] = fmaxf(red[t], red[t + o]); __syncthreads(); }
    float mx = red[0]; __syncthreads();
    float e = expf(v - mx);
    red[t] = e; __syncthreads();
    for (int o = 64; o > 0; o >>= 1) { if (t < o) red[t] += red[t + o]; __syncthreads(); }
    float sum = red[0];
    out[(long long)row * 128 + t] = (__bf16)(e / sum);
}

// GroupNorm over DH=64 per head; input phi2[(bh*128+i)*128 + l][64],
// output gn[token = ((b*128+i)*128+l)][h*64+c] (bf16).
__global__ __launch_bounds__(64) void gn_kernel(const __bf16* __restrict__ phi,
                                                __bf16* __restrict__ out) {
    __shared__ float red[64];
    int g = blockIdx.x;              // (bh*128 + i)*128 + l
    int c = threadIdx.x;
    float v = (float)phi[(long long)g * 64 + c];
    red[c] = v; __syncthreads();
    for (int o = 32; o > 0; o >>= 1) { if (c < o) red[c] += red[c + o]; __syncthreads(); }
    float m = red[0] * (1.0f / 64.0f); __syncthreads();
    float d = v - m;
    red[c] = d * d; __syncthreads();
    for (int o = 32; o > 0; o >>= 1) { if (c < o) red[c] += red[c + o]; __syncthreads(); }
    float var = red[0] * (1.0f / 64.0f);
    float inv = rsqrtf(var + 1e-6f);
    int l = g & 127, i = (g >> 7) & 127, bh = g >> 14, b = bh >> 3, h = bh & 7;
    long long tok = (long long)(b * 128 + i) * 128 + l;
    out[tok * 512 + h * 64 + c] = (__bf16)(d * inv);
}

// ------------------------------ host orchestration --------------------------

extern "C" void kernel_launch(void* const* d_in, const int* in_sizes, int n_in,
                              void* d_out, int out_size, void* d_ws, size_t ws_size,
                              hipStream_t stream) {
    (void)in_sizes; (void)n_in; (void)out_size; (void)ws_size;

    const float* u     = (const float*)d_in[0];
    const float* cos_y = (const float*)d_in[1];
    const float* sin_y = (const float*)d_in[2];
    const float* cos_x = (const float*)d_in[3];
    const float* sin_x = (const float*)d_in[4];
    /* d_in[5] scalar_cond: unused by reference */
    const float* ln1_g = (const float*)d_in[6];
    const float* ln1_b = (const float*)d_in[7];
    const float* ln2_g = (const float*)d_in[8];
    const float* ln2_b = (const float*)d_in[9];
    const float* Wv    = (const float*)d_in[10];
    const float* Wy_in = (const float*)d_in[11];
    const float* Wy1   = (const float*)d_in[12];
    const float* by1   = (const float*)d_in[13];
    const float* Wy2   = (const float*)d_in[14];
    const float* by2   = (const float*)d_in[15];
    const float* Wx_in = (const float*)d_in[16];
    const float* Wx1   = (const float*)d_in[17];
    const float* bx1   = (const float*)d_in[18];
    const float* Wx2   = (const float*)d_in[19];
    const float* bx2   = (const float*)d_in[20];
    const float* Wqk_x = (const float*)d_in[21];
    const float* Wqk_y = (const float*)d_in[22];
    const float* Wm    = (const float*)d_in[23];
    const float* bm    = (const float*)d_in[24];
    const float* Wf1   = (const float*)d_in[25];
    const float* bf1   = (const float*)d_in[26];
    const float* Wf2   = (const float*)d_in[27];
    const float* bf2   = (const float*)d_in[28];

    char* bump = (char*)d_ws;
    auto balloc = [&](size_t bytes) -> char* {
        char* r = bump; bump += (bytes + 255) & ~(size_t)255; return r;
    };
    auto cvt = [&](const float* src, int n) -> __bf16* {
        __bf16* dst = (__bf16*)balloc((size_t)n * 2);
        cvt_kernel<<<(n + 255) / 256, 256, 0, stream>>>(src, dst, n);
        return dst;
    };
    auto gemm = [&](const __bf16* A, long long sA, int adiv, int lda,
                    const __bf16* B, long long sB, int bdiv, int ldb,
                    void* C, long long sC, int cdiv, int ldc,
                    const float* bias, const float* resid,
                    int M, int N, int K, float alpha, int act, int outm, int batch) {
        GemmP prm{A, sA, adiv, lda, B, sB, bdiv, ldb, C, sC, cdiv, ldc,
                  bias, resid, M, N, K, alpha, act, outm};
        dim3 g((N + 127) / 128, (M + 127) / 128, batch);
        gemm_bf16<<<g, 256, 0, stream>>>(prm);
    };

    // bf16 weights
    __bf16* wv  = cvt(Wv,    256 * 512);
    __bf16* wyi = cvt(Wy_in, 256 * 256);
    __bf16* wy1 = cvt(Wy1,   256 * 1024);
    __bf16* wy2 = cvt(Wy2,   1024 * 256);
    __bf16* wxi = cvt(Wx_in, 256 * 256);
    __bf16* wx1 = cvt(Wx1,   256 * 1024);
    __bf16* wx2 = cvt(Wx2,   1024 * 256);
    __bf16* wqx = cvt(Wqk_x, 256 * 1024);
    __bf16* wqy = cvt(Wqk_y, 256 * 1024);
    __bf16* wm  = cvt(Wm,    512 * 256);
    __bf16* wf1 = cvt(Wf1,   256 * 1024);
    __bf16* wf2 = cvt(Wf2,   1024 * 256);

    // small scratch
    __bf16* pool_y = (__bf16*)balloc((size_t)512 * 256 * 2);
    __bf16* pool_x = (__bf16*)balloc((size_t)512 * 256 * 2);
    __bf16* t1x    = (__bf16*)balloc((size_t)512 * 256 * 2);
    __bf16* t1y    = (__bf16*)balloc((size_t)512 * 256 * 2);
    __bf16* hsx    = (__bf16*)balloc((size_t)512 * 1024 * 2);
    __bf16* hsy    = (__bf16*)balloc((size_t)512 * 1024 * 2);
    __bf16* ux     = (__bf16*)balloc((size_t)512 * 256 * 2);
    __bf16* uy     = (__bf16*)balloc((size_t)512 * 256 * 2);
    float*  qkx    = (float*) balloc((size_t)512 * 1024 * 4);
    float*  qky    = (float*) balloc((size_t)512 * 1024 * 4);
    __bf16* qxb    = (__bf16*)balloc((size_t)32 * 128 * 64 * 2);
    __bf16* kxb    = (__bf16*)balloc((size_t)32 * 128 * 64 * 2);
    __bf16* qyb    = (__bf16*)balloc((size_t)32 * 128 * 64 * 2);
    __bf16* kyb    = (__bf16*)balloc((size_t)32 * 128 * 64 * 2);
    float*  scx    = (float*) balloc((size_t)32 * 128 * 128 * 4);
    float*  scy    = (float*) balloc((size_t)32 * 128 * 128 * 4);
    __bf16* ax     = (__bf16*)balloc((size_t)32 * 128 * 128 * 2);
    __bf16* ay     = (__bf16*)balloc((size_t)32 * 128 * 128 * 2);

    // big regions (with lifetime reuse)
    __bf16* un = (__bf16*)balloc((size_t)65536 * 256 * 2);  // un -> ln2_bf
    char*   Rv = balloc((size_t)65536 * 512 * 2);           // v_bh -> phi2 -> u2(f32)
    char*   Rp = balloc((size_t)65536 * 512 * 2);           // phi1 -> gn -> hid halves

    // 1) LayerNorm1
    ln_kernel<<<65536, 256, 0, stream>>>(u, ln1_g, ln1_b, un);

    // 2) pooled means (mean commutes with the Win projection)
    pool_kernel<<<512, 256, 0, stream>>>(un, pool_y, 0);
    pool_kernel<<<512, 256, 0, stream>>>(un, pool_x, 1);

    // 3) v = un @ Wv, scattered to [b,h,y, x*64+c]
    gemm(un, 0, 1, 256, wv, 0, 1, 512, Rv, 0, 1, 512,
         nullptr, nullptr, 65536, 512, 256, 1.0f, 0, 2, 1);

    // 4) x-branch reducer + attention kernel
    gemm(pool_x, 0, 1, 256, wxi, 0, 1, 256, t1x, 0, 1, 256,
         nullptr, nullptr, 512, 256, 256, 1.0f, 0, 1, 1);
    gemm(t1x, 0, 1, 256, wx1, 0, 1, 1024, hsx, 0, 1, 1024,
         bx1, nullptr, 512, 1024, 256, 1.0f, 1, 1, 1);
    gemm(hsx, 0, 1, 1024, wx2, 0, 1, 256, ux, 0, 1, 256,
         bx2, nullptr, 512, 256, 1024, 1.0f, 0, 1, 1);
    gemm(ux, 0, 1, 256, wqx, 0, 1, 1024, qkx, 0, 1, 1024,
         nullptr, nullptr, 512, 1024, 256, 1.0f, 0, 0, 1);
    rope_kernel<<<512, 512, 0, stream>>>(qkx, cos_x, sin_x, qxb, kxb);
    gemm(qxb, 128 * 64, 1, 64, kxb, 64 * 128, 1, 128, scx, 128 * 128, 1, 128,
         nullptr, nullptr, 128, 128, 64, 1.0f / 64.0f, 0, 0, 32);
    softmax_kernel<<<4096, 128, 0, stream>>>(scx, ax);

    // 5) y-branch
    gemm(pool_y, 0, 1, 256, wyi, 0, 1, 256, t1y, 0, 1, 256,
         nullptr, nullptr, 512, 256, 256, 1.0f, 0, 1, 1);
    gemm(t1y, 0, 1, 256, wy1, 0, 1, 1024, hsy, 0, 1, 1024,
         by1, nullptr, 512, 1024, 256, 1.0f, 1, 1, 1);
    gemm(hsy, 0, 1, 1024, wy2, 0, 1, 256, uy, 0, 1, 256,
         by2, nullptr, 512, 256, 1024, 1.0f, 0, 1, 1);
    gemm(uy, 0, 1, 256, wqy, 0, 1, 1024, qky, 0, 1, 1024,
         nullptr, nullptr, 512, 1024, 256, 1.0f, 0, 0, 1);
    rope_kernel<<<512, 512, 0, stream>>>(qky, cos_y, sin_y, qyb, kyb);
    gemm(qyb, 128 * 64, 1, 64, kyb, 64 * 128, 1, 128, scy, 128 * 128, 1, 128,
         nullptr, nullptr, 128, 128, 64, 1.0f / 64.0f, 0, 0, 32);
    softmax_kernel<<<4096, 128, 0, stream>>>(scy, ay);

    // 6) einsum1: phi1[bh] (128 x 8192) = k_y[bh] @ v[bh]
    gemm(ay, (long long)128 * 128, 1, 128,
         (const __bf16*)Rv, (long long)128 * 8192, 1, 8192,
         Rp, (long long)128 * 8192, 1, 8192,
         nullptr, nullptr, 128, 8192, 128, 1.0f, 0, 1, 32);

    // 7) einsum2: phi2[bh,i] (128 x 64) = k_x[bh] @ phi1[bh,i]; 4096 batches
    gemm(ax, (long long)128 * 128, 128, 128,
         (const __bf16*)Rp, 8192, 1, 64,
         Rv, 8192, 1, 64,
         nullptr, nullptr, 128, 64, 128, 1.0f, 0, 1, 4096);

    // 8) GroupNorm (per-head DH=64) with transpose back to token-major
    gn_kernel<<<524288, 64, 0, stream>>>((const __bf16*)Rv, (__bf16*)Rp);

    // 9) u2 = gn @ Wm + bm + u  (f32, into Rv)
    gemm((const __bf16*)Rp, 0, 1, 512, wm, 0, 1, 256, Rv, 0, 1, 256,
         bm, u, 65536, 256, 512, 1.0f, 0, 0, 1);

    // 10) LayerNorm2 (reuse un region for bf16 output)
    ln_kernel<<<65536, 256, 0, stream>>>((const float*)Rv, ln2_g, ln2_b, un);

    // 11) final MLP in two half-token passes (hid reuses Rp), + residual u2
    for (int hlf = 0; hlf < 2; ++hlf) {
        const __bf16* xh  = un + (size_t)hlf * 32768 * 256;
        __bf16*       hid = (__bf16*)Rp;
        gemm(xh, 0, 1, 256, wf1, 0, 1, 1024, hid, 0, 1, 1024,
             bf1, nullptr, 32768, 1024, 256, 1.0f, 1, 1, 1);
        const float* res  = (const float*)Rv + (size_t)hlf * 32768 * 256;
        float*       outp = (float*)d_out + (size_t)hlf * 32768 * 256;
        gemm(hid, 0, 1, 1024, wf2, 0, 1, 256, outp, 0, 1, 256,
             bf2, res, 32768, 256, 1024, 1.0f, 0, 0, 1);
    }
}
